// MyTransformer_9491877724369
// MI455X (gfx1250) — matrix operand order repaired and checked by reading, not compiled, not run on hardware
//
#include <hip/hip_runtime.h>

// ---------------------------------------------------------------------------
// Transformer enc/dec forward for MI455X (gfx1250, wave32).
// All GEMMs -> V_WMMA_F32_16X16X32_BF16 (bf16 A/B, f32 accum).
// bf16 conversion happens exactly once per value (pack kernels / epilogues);
// GEMM inner loop is pure 128-bit loads + WMMA, 32x64 tile per wave
// (8 WMMAs per 12 vector loads per k-step).
// ---------------------------------------------------------------------------

#define S_LEN    2048
#define D_DIM    512
#define H_HEADS  8
#define DH_DIM   64
#define L_LAYERS 8
#define V_SIZE   32000

typedef __attribute__((ext_vector_type(16))) __bf16 v16bf;
typedef __attribute__((ext_vector_type(8)))  __bf16 v8bf;
typedef __attribute__((ext_vector_type(8)))  float  v8f;

// ---------------------------------------------------------------------------
// Fragment loaders (bf16 operands, contiguous 16B vector loads)
// ---------------------------------------------------------------------------

// A fragment: 16x32 (MxK), row-major bf16 A.
// Lane L<16: row M=L, K = 0..7,16..23 ; lane L>=16: row M=L-16, K = 8..15,24..31.
__device__ __forceinline__ v16bf load_a16(const __bf16* __restrict__ A, int lda,
                                          int row0, int k0, int lane) {
  const __bf16* p = A + (size_t)(row0 + (lane & 15)) * lda + k0 + ((lane >> 4) << 3);
  v8bf lo = *(const v8bf*)p;        // K 0..7   (+kofs)
  v8bf hi = *(const v8bf*)(p + 16); // K 16..23 (+kofs)
  return __builtin_shufflevector(lo, hi, 0, 1, 2, 3, 4, 5, 6, 7,
                                 8, 9, 10, 11, 12, 13, 14, 15);
}

// B fragment: 32x16 (KxN) where logical B[k][n] = Bt[n][k], Bt row-major (ld=ldb).
// Lanes 0-15: K = 0..7,16..23 at N=lane; lanes 16-31: K = 8..15,24..31 at N=lane-16.
// Each lane reads two runs of 8 bf16 -> two 128-bit loads.
__device__ __forceinline__ v16bf load_bt16(const __bf16* __restrict__ Bt, int ldb,
                                           int k0, int col0, int lane) {
  const __bf16* p = Bt + (size_t)(col0 + (lane & 15)) * ldb + k0 + ((lane >> 4) << 3);
  v8bf lo = *(const v8bf*)p;
  v8bf hi = *(const v8bf*)(p + 16);
  return __builtin_shufflevector(lo, hi, 0, 1, 2, 3, 4, 5, 6, 7,
                                 8, 9, 10, 11, 12, 13, 14, 15);
}

// ---------------------------------------------------------------------------
// GEMM: C[M,N] = act(A[M,K] @ Bt[N,K]^T + bias) (+ fp32 residual)
// Outputs: fp32 (Cf, ld=ldcf) and/or bf16 (Cb, ld=ldcb; TRANSC -> Cb[n*ldcb+m]).
// Res (fp32) uses ldcf. Batched over blockIdx.z.
// Wave -> 32x64 tile (2 row-tiles x 4 col-tiles, 8 accumulators);
// block = 4 waves -> 128x64. Requires M % 128 == 0 (always 2048 here).
// ---------------------------------------------------------------------------
template <bool RELU, bool ADDRES, bool OUTF32, bool OUTBF16, bool TRANSC>
__global__ __launch_bounds__(128) void wmma_gemm(
    const __bf16* __restrict__ A, const __bf16* __restrict__ Bt,
    const float* __restrict__ bias, const float* __restrict__ Res,
    float* __restrict__ Cf, __bf16* __restrict__ Cb,
    int M, int N, int K, int lda, int ldb, int ldcf, int ldcb,
    long long sA, long long sB, long long sBias, long long sCf, long long sCb) {
  long long z = blockIdx.z;
  A += z * sA;
  Bt += z * sB;
  if (bias) bias += z * sBias;
  if (OUTF32) Cf += z * sCf;
  if (OUTBF16) Cb += z * sCb;

  int lane = threadIdx.x & 31;
  int wave = threadIdx.x >> 5;
  int row0 = (blockIdx.y * 4 + wave) * 32;
  int col0 = blockIdx.x * 64;
  if (row0 >= M || col0 >= N) return;

  v8f acc[2][4] = {};
  for (int k0 = 0; k0 < K; k0 += 32) {
    v16bf a0 = load_a16(A, lda, row0, k0, lane);
    v16bf a1 = load_a16(A, lda, row0 + 16, k0, lane);
#pragma unroll
    for (int j = 0; j < 4; ++j) {
      v16bf b = load_bt16(Bt, ldb, k0, col0 + j * 16, lane);
      acc[0][j] = __builtin_amdgcn_wmma_f32_16x16x32_bf16(
          false, a0, false, b, (short)0, acc[0][j], false, false);
      acc[1][j] = __builtin_amdgcn_wmma_f32_16x16x32_bf16(
          false, a1, false, b, (short)0, acc[1][j], false, false);
    }
  }

  // C/D layout: VGPR r holds M = rt*16 + r + 8*(lane>=16), N = col0 + (lane&15).
  int hi = lane >> 4;
  int nl = lane & 15;
#pragma unroll
  for (int rt = 0; rt < 2; ++rt) {
#pragma unroll
    for (int j = 0; j < 4; ++j) {
      int cb = col0 + j * 16 + nl;
      float bv = bias ? bias[cb] : 0.0f;
#pragma unroll
      for (int r = 0; r < 8; ++r) {
        int rr = row0 + rt * 16 + r + hi * 8;
        float v = acc[rt][j][r] + bv;
        if (RELU) v = fmaxf(v, 0.0f);
        if (ADDRES) v += Res[(size_t)rr * ldcf + cb];
        if (OUTF32) Cf[(size_t)rr * ldcf + cb] = v;
        if (OUTBF16) {
          if (TRANSC) Cb[(size_t)cb * ldcb + rr] = (__bf16)v;
          else        Cb[(size_t)rr * ldcb + cb] = (__bf16)v;
        }
      }
    }
  }
}

// ---------------------------------------------------------------------------
// Row softmax over scores[H*S, S] (scale 1/sqrt(DH), optional causal mask).
// Reads fp32 in registers, writes normalized attention IN PLACE as bf16
// (row stride 2*S bf16). Masked tail written as bf16 zeros.
// ---------------------------------------------------------------------------
__global__ __launch_bounds__(256) void softmax_mask_bf16(float* __restrict__ sc,
                                                         int S, int causal,
                                                         float scale) {
  __shared__ float red[256];
  int row = blockIdx.x;          // h*S + q
  int q   = row % S;
  float*  p  = sc + (size_t)row * S;
  __bf16* pb = (__bf16*)p;       // bf16 view of the same row
  int nv  = causal ? (q + 1) : S;
  int tid = threadIdx.x;

  float m = -3.4e38f;
  for (int s = tid; s < nv; s += 256) m = fmaxf(m, p[s] * scale);
  red[tid] = m;
  __syncthreads();
  for (int o = 128; o > 0; o >>= 1) {
    if (tid < o) red[tid] = fmaxf(red[tid], red[tid + o]);
    __syncthreads();
  }
  m = red[0];
  __syncthreads();

  float vals[8];  // S/256 = 8 max per thread
  int   cnt = 0;
  float sum = 0.0f;
  for (int s = tid; s < nv; s += 256) {
    float e = __expf(p[s] * scale - m);
    vals[cnt++] = e;
    sum += e;
  }
  red[tid] = sum;
  __syncthreads();
  for (int o = 128; o > 0; o >>= 1) {
    if (tid < o) red[tid] += red[tid + o];
    __syncthreads();
  }
  float inv = 1.0f / red[0];
  // All fp32 reads of this row completed before the barriers above; safe to
  // overwrite with bf16 now.
  cnt = 0;
  for (int s = tid; s < nv; s += 256) pb[s] = (__bf16)(vals[cnt++] * inv);
  for (int s = nv + tid; s < S; s += 256) pb[s] = (__bf16)0.0f;
}

// out[s,:] = emb[ids[s],:] + PE(s,:) -> fp32 + bf16 mirror. Faithful PE:
// columns 0..509, sin even / cos odd with the same exponent 2j/D; 510,511 = 0.
__global__ __launch_bounds__(512) void embed_pos(const int* __restrict__ ids,
                                                 const float* __restrict__ emb,
                                                 float* __restrict__ out,
                                                 __bf16* __restrict__ out16) {
  int s = blockIdx.x;
  int d = threadIdx.x;  // 0..511
  float pe = 0.0f;
  if (d < 2 * ((D_DIM - 1) / 2)) {  // d < 510
    int j = d >> 1;
    float invf = __expf(-(2.0f * (float)j / (float)D_DIM) * 9.210340371976184f);
    float ang  = (float)s * invf;
    pe = (d & 1) ? __cosf(ang) : __sinf(ang);
  }
  float v = emb[(size_t)ids[s] * D_DIM + d] + pe;
  out[(size_t)s * D_DIM + d]   = v;
  out16[(size_t)s * D_DIM + d] = (__bf16)v;
}

// Transpose-convert fp32 [K][N] -> bf16 [N][K], batched over blockIdx.z.
__global__ __launch_bounds__(256) void transpose_cvt(const float* __restrict__ in,
                                                     __bf16* __restrict__ out,
                                                     int K, int N) {
  long long z = blockIdx.z;
  in  += z * (long long)K * N;
  out += z * (long long)K * N;
  int idx = blockIdx.x * 256 + threadIdx.x;
  if (idx >= K * N) return;
  int k = idx / N, n = idx % N;
  out[(size_t)n * K + k] = (__bf16)in[idx];
}

// ---------------------------------------------------------------------------
// Host-side orchestration
// ---------------------------------------------------------------------------

enum GemmMode { M_BF16, M_BF16T, M_F32, M_BF16RES, M_BF16RELU, M_BOTH };

static void gemm(hipStream_t st, GemmMode mode, const __bf16* A, const __bf16* Bt,
                 const float* bias, const float* Res, float* Cf, __bf16* Cb,
                 int M, int N, int K, int lda, int ldb, int ldcf, int ldcb,
                 long long sA, long long sB, long long sBias, long long sCf,
                 long long sCb, int batch) {
  dim3 g(N / 64, M / 128, batch), b(128);
  switch (mode) {
    case M_BF16:
      wmma_gemm<false, false, false, true, false><<<g, b, 0, st>>>(
          A, Bt, bias, Res, Cf, Cb, M, N, K, lda, ldb, ldcf, ldcb, sA, sB, sBias, sCf, sCb);
      break;
    case M_BF16T:
      wmma_gemm<false, false, false, true, true><<<g, b, 0, st>>>(
          A, Bt, bias, Res, Cf, Cb, M, N, K, lda, ldb, ldcf, ldcb, sA, sB, sBias, sCf, sCb);
      break;
    case M_F32:
      wmma_gemm<false, false, true, false, false><<<g, b, 0, st>>>(
          A, Bt, bias, Res, Cf, Cb, M, N, K, lda, ldb, ldcf, ldcb, sA, sB, sBias, sCf, sCb);
      break;
    case M_BF16RES:
      wmma_gemm<false, true, false, true, false><<<g, b, 0, st>>>(
          A, Bt, bias, Res, Cf, Cb, M, N, K, lda, ldb, ldcf, ldcb, sA, sB, sBias, sCf, sCb);
      break;
    case M_BF16RELU:
      wmma_gemm<true, false, false, true, false><<<g, b, 0, st>>>(
          A, Bt, bias, Res, Cf, Cb, M, N, K, lda, ldb, ldcf, ldcb, sA, sB, sBias, sCf, sCb);
      break;
    case M_BOTH:
      wmma_gemm<false, false, true, true, false><<<g, b, 0, st>>>(
          A, Bt, bias, Res, Cf, Cb, M, N, K, lda, ldb, ldcf, ldcb, sA, sB, sBias, sCf, sCb);
      break;
  }
}

struct Ws {
  float *xe, *yd, *scores;                       // fp32
  __bf16 *xe16, *yd16, *q16, *k16, *vT16;        // bf16 activations
  __bf16 *ctx16, *hb16, *fb16;
  __bf16 *wqkvT[3], *woT[3], *w1T[3], *w2T[3];   // bf16 transposed weights
  __bf16 *cWqkvT, *cWoT, *outWT;
};

// One encoder/decoder block: cur = FFN(attn(cur) + cur)
static void attn_ffn_layer(hipStream_t st, const Ws& w, float* cur, __bf16* cur16,
                           const __bf16* WqkvT, const float* bqkv,
                           const __bf16* WoT, const float* bo,
                           const __bf16* W1T, const float* b1,
                           const __bf16* W2T, const float* b2, bool causal) {
  const int S = S_LEN, D = D_DIM, H = H_HEADS, DH = DH_DIM;
  // q,k (t=0,1; batch 16): qkv16[t][h][S][DH] (q16/k16 contiguous)
  gemm(st, M_BF16, cur16, WqkvT, bqkv, nullptr, nullptr, w.q16,
       S, DH, D, D, D, 0, DH, 0, (long long)D * DH, DH, 0, (long long)S * DH, 16);
  // v (t=2; batch 8): transposed write -> vT16[h][DH][S]
  gemm(st, M_BF16T, cur16, WqkvT + (size_t)2 * H * D * DH, bqkv + 2 * H * DH,
       nullptr, nullptr, w.vT16,
       S, DH, D, D, D, 0, S, 0, (long long)D * DH, DH, 0, (long long)S * DH, 8);
  // scores[h] = q_h @ k_h^T (Bt = k16, naturally [s][dh])
  gemm(st, M_F32, w.q16, w.k16, nullptr, nullptr, w.scores, nullptr,
       S, S, DH, DH, DH, S, 0,
       (long long)S * DH, (long long)S * DH, 0, (long long)S * S, 0, 8);
  softmax_mask_bf16<<<H * S, 256, 0, st>>>(w.scores, S, causal ? 1 : 0, 0.125f);
  // ctx[:, h*64:+64] = att_h @ v_h  (A = bf16 att in-place, row stride 2S)
  gemm(st, M_BF16, (const __bf16*)w.scores, w.vT16, nullptr, nullptr, nullptr,
       w.ctx16, S, DH, S, 2 * S, S, 0, D,
       (long long)2 * S * S, (long long)S * DH, 0, 0, 64LL, 8);
  // h = ctx @ Wo + bo + cur (fp32 residual) -> bf16
  gemm(st, M_BF16RES, w.ctx16, WoT, bo, cur, nullptr, w.hb16,
       S, D, D, D, D, D, D, 0, 0, 0, 0, 0, 1);
  // FFN
  gemm(st, M_BF16RELU, w.hb16, W1T, b1, nullptr, nullptr, w.fb16,
       S, D, D, D, D, 0, D, 0, 0, 0, 0, 0, 1);
  gemm(st, M_BOTH, w.fb16, W2T, b2, nullptr, cur, cur16,
       S, D, D, D, D, D, D, 0, 0, 0, 0, 0, 1);
}

extern "C" void kernel_launch(void* const* d_in, const int* in_sizes, int n_in,
                              void* d_out, int out_size, void* d_ws, size_t ws_size,
                              hipStream_t stream) {
  (void)in_sizes; (void)n_in; (void)out_size; (void)ws_size;
  const int S = S_LEN, D = D_DIM, H = H_HEADS, DH = DH_DIM, L = L_LAYERS;

  // setup_inputs() dict order
  const int*   x_ids      = (const int*)d_in[0];
  const int*   y_ids      = (const int*)d_in[1];
  const float* emb_en     = (const float*)d_in[2];
  const float* emb_de     = (const float*)d_in[3];
  const float* cross_Wqkv = (const float*)d_in[4];
  const float* cross_bqkv = (const float*)d_in[5];
  const float* cross_Wo   = (const float*)d_in[6];
  const float* cross_bo   = (const float*)d_in[7];
  const float* out_W      = (const float*)d_in[8];
  const float* out_b      = (const float*)d_in[9];
  const float* stk[3][8];  // {Wqkv,bqkv,Wo,bo,W1,b1,W2,b2} x {enc,dec,ac}
  for (int s = 0; s < 3; ++s)
    for (int t = 0; t < 8; ++t) stk[s][t] = (const float*)d_in[10 + s * 8 + t];

  // ---- workspace layout ----
  const size_t SD = (size_t)S * D;  // 1,048,576
  float* Wf = (float*)d_ws;
  Ws w;
  w.xe = Wf;               // SD f32
  w.yd = Wf + SD;          // SD f32
  w.scores = Wf + 2 * SD;  // 32*SD f32 (H*S*S); also aliased as bf16 attention
  __bf16* bf = (__bf16*)(Wf + 34 * SD);
  size_t o = 0;
  w.xe16 = bf + o; o += SD;
  w.yd16 = bf + o; o += SD;
  w.q16  = bf + o; o += SD;   // [H][S][DH]
  w.k16  = bf + o; o += SD;   // [H][S][DH] (contiguous after q16)
  w.vT16 = bf + o; o += SD;   // [H][DH][S]
  w.ctx16 = bf + o; o += SD;
  w.hb16  = bf + o; o += SD;
  w.fb16  = bf + o; o += SD;
  for (int s = 0; s < 3; ++s) {
    w.wqkvT[s] = bf + o; o += (size_t)L * 3 * H * D * DH;  // 6*SD
    w.woT[s]   = bf + o; o += (size_t)L * D * D;           // 2*SD
    w.w1T[s]   = bf + o; o += (size_t)L * D * D;
    w.w2T[s]   = bf + o; o += (size_t)L * D * D;
  }
  w.cWqkvT = bf + o; o += (size_t)3 * H * D * DH;
  w.cWoT   = bf + o; o += (size_t)D * D;
  w.outWT  = bf + o; o += (size_t)D * V_SIZE;

  // ---- one-time (per call) weight transpose+convert to bf16 ----
  auto tc = [&](const float* in, __bf16* out, int K, int N, int batch) {
    long long kn = (long long)K * N;
    dim3 g((unsigned)((kn + 255) / 256), 1, batch);
    transpose_cvt<<<g, 256, 0, stream>>>(in, out, K, N);
  };
  for (int s = 0; s < 3; ++s) {
    tc(stk[s][0], w.wqkvT[s], D, DH, L * 3 * H);
    tc(stk[s][2], w.woT[s], D, D, L);
    tc(stk[s][4], w.w1T[s], D, D, L);
    tc(stk[s][6], w.w2T[s], D, D, L);
  }
  tc(cross_Wqkv, w.cWqkvT, D, DH, 3 * H);
  tc(cross_Wo, w.cWoT, D, D, 1);
  tc(out_W, w.outWT, D, V_SIZE, 1);

  // ---- embeddings + positional encoding ----
  embed_pos<<<S, D, 0, stream>>>(x_ids, emb_en, w.xe, w.xe16);
  embed_pos<<<S, D, 0, stream>>>(y_ids, emb_de, w.yd, w.yd16);

  const size_t sWqkv = (size_t)3 * H * D * DH;
  const size_t sbqkv = (size_t)3 * H * DH;
  const size_t sMat  = (size_t)D * D;
  const size_t sVec  = (size_t)D;

  // ---- encoder stack (non-causal) on x ----
  for (int l = 0; l < L; ++l)
    attn_ffn_layer(stream, w, w.xe, w.xe16,
                   w.wqkvT[0] + l * sWqkv, stk[0][1] + l * sbqkv,
                   w.woT[0] + l * sMat, stk[0][3] + l * sVec,
                   w.w1T[0] + l * sMat, stk[0][5] + l * sVec,
                   w.w2T[0] + l * sMat, stk[0][7] + l * sVec, false);

  // ---- decoder stack (causal) on y ----
  for (int l = 0; l < L; ++l)
    attn_ffn_layer(stream, w, w.yd, w.yd16,
                   w.wqkvT[1] + l * sWqkv, stk[1][1] + l * sbqkv,
                   w.woT[1] + l * sMat, stk[1][3] + l * sVec,
                   w.w1T[1] + l * sMat, stk[1][5] + l * sVec,
                   w.w2T[1] + l * sMat, stk[1][7] + l * sVec, true);

  // ---- cross attention: Q from y (dec out), K/V from x (enc out); no mask,
  // no residual. Result z overwrites x (enc acts fully consumed). ----
  gemm(stream, M_BF16, w.yd16, w.cWqkvT, cross_bqkv, nullptr, nullptr, w.q16,
       S, DH, D, D, D, 0, DH, 0, (long long)D * DH, DH, 0, (long long)S * DH, 8);
  gemm(stream, M_BF16, w.xe16, w.cWqkvT + (size_t)H * D * DH, cross_bqkv + H * DH,
       nullptr, nullptr, w.k16,
       S, DH, D, D, D, 0, DH, 0, (long long)D * DH, DH, 0, (long long)S * DH, 8);
  gemm(stream, M_BF16T, w.xe16, w.cWqkvT + (size_t)2 * H * D * DH,
       cross_bqkv + 2 * H * DH, nullptr, nullptr, w.vT16,
       S, DH, D, D, D, 0, S, 0, (long long)D * DH, DH, 0, (long long)S * DH, 8);
  gemm(stream, M_F32, w.q16, w.k16, nullptr, nullptr, w.scores, nullptr,
       S, S, DH, DH, DH, S, 0,
       (long long)S * DH, (long long)S * DH, 0, (long long)S * S, 0, 8);
  softmax_mask_bf16<<<H * S, 256, 0, stream>>>(w.scores, S, 0, 0.125f);
  gemm(stream, M_BF16, (const __bf16*)w.scores, w.vT16, nullptr, nullptr, nullptr,
       w.ctx16, S, DH, S, 2 * S, S, 0, D,
       (long long)2 * S * S, (long long)S * DH, 0, 0, 64LL, 8);
  gemm(stream, M_BOTH, w.ctx16, w.cWoT, cross_bo, nullptr, w.xe, w.xe16,
       S, D, D, D, D, D, D, 0, 0, 0, 0, 0, 1);

  // ---- "ac" stack (non-causal) on z ----
  for (int l = 0; l < L; ++l)
    attn_ffn_layer(stream, w, w.xe, w.xe16,
                   w.wqkvT[2] + l * sWqkv, stk[2][1] + l * sbqkv,
                   w.woT[2] + l * sMat, stk[2][3] + l * sVec,
                   w.w1T[2] + l * sMat, stk[2][5] + l * sVec,
                   w.w2T[2] + l * sMat, stk[2][7] + l * sVec, false);

  // ---- logits: [S, V] = z @ out_W + out_b ----
  gemm(stream, M_F32, w.xe16, w.outWT, out_b, nullptr, (float*)d_out, nullptr,
       S, V_SIZE, D, D, D, V_SIZE, 0, 0, 0, 0, 0, 0, 1);
}
